// SSIMLoss_31851477467721
// MI455X (gfx1250) — matrix operand order repaired and checked by reading, not compiled, not run on hardware
//
#include <hip/hip_runtime.h>
#include <hip/hip_bf16.h>

typedef __attribute__((ext_vector_type(16))) _Float16 v16h;
typedef __attribute__((ext_vector_type(8)))  float    v8f;

#define IMG_H 512
#define IMG_W 512
#define NIMG  48            // B*C = 16*3
#define TILES 32            // 512/16 per dim
#define WAVES_PER_BLOCK 8
#define KS   11
#define RAD  5
#define IN_ROWS 26          // 16 + 2*RAD halo rows
#define IN_STRIDE 33        // padded row stride (floats) to dodge bank conflicts
#define T_STRIDE 34         // padded row stride (halfs) for t staging
#define C1F 1e-4f
#define C2F 9e-4f

__shared__ float    sX[WAVES_PER_BLOCK][IN_ROWS * IN_STRIDE];
__shared__ float    sY[WAVES_PER_BLOCK][IN_ROWS * IN_STRIDE];
__shared__ _Float16 sT[WAVES_PER_BLOCK][16 * T_STRIDE];

__global__ __launch_bounds__(256) void ssim_main(
    const float* __restrict__ x, const float* __restrict__ y,
    const float* __restrict__ kern, double* __restrict__ acc)
{
    const int tid  = threadIdx.x;
    const int lane = tid & 31;
    const int wave = tid >> 5;
    const int hw   = lane >> 4;      // half-wave index
    const int ln16 = lane & 15;

    const int gw      = blockIdx.x * WAVES_PER_BLOCK + wave;  // tile id
    const int img     = gw >> 10;                              // / (32*32)
    const int tileIdx = gw & 1023;
    const int r0 = (tileIdx >> 5) * 16;
    const int c0 = (tileIdx & 31) * 16;

    const float* xi = x + (size_t)img * IMG_H * IMG_W;
    const float* yi = y + (size_t)img * IMG_H * IMG_W;

    // ---- recover separable 1D weights from the rank-1 2D kernel ----
    float w1[KS];
    {
        float g5 = sqrtf(kern[5 * KS + 5]);     // g1[5]
        #pragma unroll
        for (int i = 0; i < KS; ++i) w1[i] = kern[i * KS + 5] / g5;
    }

    // ---- banded weight matrix as WMMA A operand (16x32 f16) ----
    // A[m][k] = w1[k-m] for 0 <= k-m <= 10 else 0.  Contiguous k order, not the ISA table order:
    // lane<16: v0..v7 hold K=0..15; lane>=16: K=16..31 (matches B operands)
    v16h aw;
    #pragma unroll
    for (int s = 0; s < 16; ++s) {
        int K = 16 * hw + s;
        int d = K - ln16;
        aw[s] = (d >= 0 && d < KS) ? (_Float16)w1[d] : (_Float16)0.0f;
    }

    // ---- per-wave cooperative halo load: 26 rows x 32 cols, zero-padded ----
    float* lx = sX[wave];
    float* ly = sY[wave];
    #pragma unroll 2
    for (int i = 0; i < IN_ROWS; ++i) {
        int rr = r0 - RAD + i;
        int cc = c0 - RAD + lane;
        bool ok = (rr >= 0) && (rr < IMG_H) && (cc >= 0) && (cc < IMG_W);
        size_t gidx = (size_t)rr * IMG_W + cc;
        lx[i * IN_STRIDE + lane] = ok ? xi[gidx] : 0.0f;
        ly[i * IN_STRIDE + lane] = ok ? yi[gidx] : 0.0f;
    }
    asm volatile("s_wait_dscnt 0x0" ::: "memory");

    _Float16* tb = sT[wave];
    v8f d2[5];

    #pragma unroll
    for (int q = 0; q < 5; ++q) {
        // ===== vertical pass: t[m][c] = sum_k w[k-m] * In[k][c], two 16-col strips
        #pragma unroll
        for (int tc = 0; tc < 2; ++tc) {
            const int col = ln16 + 16 * tc;
            // B operand (32x16 f16): lane<16 holds K=0..15, lane>=16 K=16..31
            v16h bv;
            #pragma unroll
            for (int s = 0; s < 16; ++s) {
                int K  = 16 * hw + s;
                int kr = (K < IN_ROWS) ? K : (IN_ROWS - 1); // A-row is 0 there anyway
                float fx = lx[kr * IN_STRIDE + col];
                float fy = ly[kr * IN_STRIDE + col];
                float v;
                switch (q) {
                    case 0:  v = fx;      break;
                    case 1:  v = fy;      break;
                    case 2:  v = fx * fx; break;
                    case 3:  v = fy * fy; break;
                    default: v = fx * fy; break;
                }
                bv[s] = (_Float16)v;
            }
            v8f cz = {};
            v8f t = __builtin_amdgcn_wmma_f32_16x16x32_f16(
                        false, aw, false, bv, (short)0, cz, false, false);
            // C/D layout: element (M = v + 8*hw, N = ln16); stage t as f16 row-major
            #pragma unroll
            for (int v = 0; v < 8; ++v) {
                int M = v + 8 * hw;
                tb[M * T_STRIDE + ln16 + 16 * tc] = (_Float16)t[v];
            }
        }
        asm volatile("s_wait_dscnt 0x0" ::: "memory");

        // ===== horizontal pass (transposed output): D2[c][r] = sum_j w[j-c]*t[r][j]
        // B operand = t^T: lane's N = r = ln16, K = j = 16*hw + s  -> contiguous row read
        v16h bt;
        {
            const int r = ln16;
            #pragma unroll
            for (int s = 0; s < 16; ++s)
                bt[s] = tb[r * T_STRIDE + 16 * hw + s];
        }
        v8f cz = {};
        d2[q] = __builtin_amdgcn_wmma_f32_16x16x32_f16(
                    false, aw, false, bt, (short)0, cz, false, false);
        asm volatile("s_wait_dscnt 0x0" ::: "memory"); // t reuse next q
    }

    // ---- pointwise SSIM + reduction (layout-agnostic: pure sum) ----
    float lsum = 0.0f;
    #pragma unroll
    for (int v = 0; v < 8; ++v) {
        float ux  = d2[0][v], uy  = d2[1][v];
        float uxx = d2[2][v], uyy = d2[3][v], uxy = d2[4][v];
        float vx  = uxx - ux * ux;
        float vy  = uyy - uy * uy;
        float vxy = uxy - ux * uy;
        float num = (2.0f * ux * uy + C1F) * (2.0f * vxy + C2F);
        float den = (ux * ux + uy * uy + C1F) * (vx + vy + C2F);
        lsum += num / (den + 1e-12f);
    }
    #pragma unroll
    for (int off = 16; off > 0; off >>= 1)
        lsum += __shfl_down(lsum, off, 32);
    if (lane == 0)
        atomicAdd(acc, (double)lsum);
}

__global__ void ssim_zero(double* acc) { *acc = 0.0; }

__global__ void ssim_finalize(const double* __restrict__ acc, float* __restrict__ out) {
    const double n = (double)NIMG * IMG_H * IMG_W; // 12,582,912
    out[0] = 1.0f - (float)(*acc / n);
}

extern "C" void kernel_launch(void* const* d_in, const int* in_sizes, int n_in,
                              void* d_out, int out_size, void* d_ws, size_t ws_size,
                              hipStream_t stream)
{
    const float* x = (const float*)d_in[0];
    const float* y = (const float*)d_in[1];
    const float* k = (const float*)d_in[2];
    double* acc = (double*)d_ws;

    ssim_zero<<<1, 1, 0, stream>>>(acc);

    const int nTiles = NIMG * TILES * TILES;          // 49152 waves
    const int blocks = nTiles / WAVES_PER_BLOCK;      // 6144 blocks of 256
    ssim_main<<<blocks, 256, 0, stream>>>(x, y, k, acc);

    ssim_finalize<<<1, 1, 0, stream>>>(acc, (float*)d_out);
}
